// SelfAttention_26680336842863
// MI455X (gfx1250) — matrix operand order repaired and checked by reading, not compiled, not run on hardware
//
#include <hip/hip_runtime.h>
#include <hip/hip_bf16.h>
#include <stdint.h>

typedef __attribute__((ext_vector_type(16))) __bf16 v16bf;
typedef __attribute__((ext_vector_type(8)))  float  v8f;

#define BATCH 4
#define CH    256
#define CQ    32
#define NPIX  4096

__device__ __forceinline__ unsigned short f2bf(float f) {
  union { float f; unsigned int u; } v; v.f = f;
  unsigned int u = v.u;
  u += 0x7FFFu + ((u >> 16) & 1u);   // round-to-nearest-even
  return (unsigned short)(u >> 16);
}

union Frag {
  uint4 q[2];
  v16bf v;
};

// Async DMA: global -> LDS, 16B per lane, tracked by ASYNCcnt.
// LDS address = low 32 bits of the generic shared-aperture pointer (ISA 10.2).
__device__ __forceinline__ void async_b128(unsigned short* lds,
                                           const unsigned short* g) {
  unsigned loff = (unsigned)(size_t)lds;
  asm volatile("global_load_async_to_lds_b128 %0, %1, off"
               :: "v"(loff), "v"(g) : "memory");
}
__device__ __forceinline__ void wait_async_le9() {
  asm volatile("s_wait_asynccnt 0x9" ::: "memory");
}
__device__ __forceinline__ void wait_async_0() {
  asm volatile("s_wait_asynccnt 0x0" ::: "memory");
}

// ---------------------------------------------------------------------------
// Kernel 1: fused QKV projection.  Y[o,n] = sum_c W[o,c] * X[c,n] + bias[o]
// Concatenated outputs: o in [0,32)=Q, [32,64)=K, [64,320)=V.
// grid = (NPIX/64, 10, BATCH), block = 256 (8 waves).  Each block: 32 o x 64 n.
// ---------------------------------------------------------------------------
#define WSTR 264   // bf16 elems per LDS row (16B-aligned, bank-skewed)
#define XSTR 264

__global__ __launch_bounds__(256) void proj_kernel(
    const float* __restrict__ x,
    const float* __restrict__ Wq, const float* __restrict__ bq,
    const float* __restrict__ Wk, const float* __restrict__ bk,
    const float* __restrict__ Wv, const float* __restrict__ bv,
    unsigned short* __restrict__ Qb,   // [B][N][32]  bf16
    unsigned short* __restrict__ Kb,   // [B][N][32]  bf16
    unsigned short* __restrict__ Vb)   // [B][256][N] bf16 (c-major!)
{
  __shared__ unsigned short Wlds[32 * WSTR];
  __shared__ unsigned short Xlds[64 * XSTR];
  __shared__ float blds[32];

  const int t    = threadIdx.x;
  const int lane = t & 31;
  const int wave = t >> 5;
  const int n0   = blockIdx.x * 64;
  const int ob   = blockIdx.y;          // 0..9, 32 outputs each
  const int b    = blockIdx.z;

  const float* Wsrc;
  if (ob == 0)      Wsrc = Wq;
  else if (ob == 1) Wsrc = Wk;
  else              Wsrc = Wv + (size_t)(ob - 2) * 32 * CH;
  for (int o = 0; o < 32; ++o)
    Wlds[o * WSTR + t] = f2bf(Wsrc[(size_t)o * CH + t]);
  if (t < 32)
    blds[t] = (ob == 0) ? bq[t] : (ob == 1) ? bk[t] : bv[(ob - 2) * 32 + t];

  for (int cc = 0; cc < CH; cc += 4) {
    int c = cc + (t >> 6);
    int n = t & 63;
    Xlds[n * XSTR + c] = f2bf(x[((size_t)b * CH + c) * NPIX + n0 + n]);
  }
  __syncthreads();

  const int half = (lane >= 16);
  const int l16  = lane & 15;
  const int k0   = half ? 8 : 0;

  const int os = wave >> 2;
  const int ns = wave & 3;

  v8f acc;
#pragma unroll
  for (int i = 0; i < 8; ++i) acc[i] = 0.0f;

#pragma unroll
  for (int kc = 0; kc < CH; kc += 32) {
    Frag a, bf;
    const unsigned short* wp = &Wlds[(os * 16 + l16) * WSTR + kc + k0];
    a.q[0] = *(const uint4*)&wp[0];
    a.q[1] = *(const uint4*)&wp[16];
    const unsigned short* xp = &Xlds[(ns * 16 + l16) * XSTR + kc + k0];
    bf.q[0] = *(const uint4*)&xp[0];
    bf.q[1] = *(const uint4*)&xp[16];
    acc = __builtin_amdgcn_wmma_f32_16x16x32_bf16(false, a.v, false, bf.v,
                                                  (short)0, acc, false, false);
  }

  const int obase = os * 16 + (half ? 8 : 0);
#pragma unroll
  for (int r = 0; r < 8; ++r) acc[r] += blds[obase + r];

  const int n = n0 + ns * 16 + l16;
  if (ob < 2) {
    uint4 pk;
    pk.x = (unsigned)f2bf(acc[0]) | ((unsigned)f2bf(acc[1]) << 16);
    pk.y = (unsigned)f2bf(acc[2]) | ((unsigned)f2bf(acc[3]) << 16);
    pk.z = (unsigned)f2bf(acc[4]) | ((unsigned)f2bf(acc[5]) << 16);
    pk.w = (unsigned)f2bf(acc[6]) | ((unsigned)f2bf(acc[7]) << 16);
    unsigned short* dst = (ob == 0) ? Qb : Kb;
    *(uint4*)&dst[((size_t)b * NPIX + n) * CQ + obase] = pk;
  } else {
    const int cbase = (ob - 2) * 32 + obase;
#pragma unroll
    for (int r = 0; r < 8; ++r)
      Vb[((size_t)b * CH + cbase + r) * NPIX + n] = f2bf(acc[r]);
  }
}

// ---------------------------------------------------------------------------
// Kernel 2: flash attention + residual, double-buffered async K/V staging.
// grid = (NPIX/128, BATCH), block = 256 (8 waves, 16 query rows per wave).
// ---------------------------------------------------------------------------
#define KSTR 40
#define VSTR 72
#define PSTR 72

__global__ __launch_bounds__(256) void attn_kernel(
    const unsigned short* __restrict__ Qb,
    const unsigned short* __restrict__ Kb,
    const unsigned short* __restrict__ Vb,
    const float* __restrict__ x,
    const float* __restrict__ gamma,
    float* __restrict__ out)
{
  __shared__ unsigned short Klds[2][64 * KSTR];       // [m][c]  2 x  5 KB
  __shared__ unsigned short Vlds[2][CH * VSTR];       // [c][m]  2 x 36 KB
  __shared__ unsigned short Plds[8][16 * PSTR];       // per-wave P     18 KB

  const int t    = threadIdx.x;
  const int lane = t & 31;
  const int wave = t >> 5;
  const int b    = blockIdx.y;
  const int qw   = blockIdx.x * 128 + wave * 16;
  const int half = (lane >= 16);
  const int l16  = lane & 15;
  const int k0   = half ? 8 : 0;

  // per-thread staging coordinates (9 x b128 per tile: 1 K + 8 V)
  const int sm  = t >> 2,      scg = (t & 3) * 8;     // K tile
  const size_t kgbase = ((size_t)b * NPIX + sm) * CQ + scg;

  // Q fragment: A-layout, resident in registers for whole kernel
  Frag qf;
  {
    const unsigned short* qp = Qb + ((size_t)b * NPIX + qw + l16) * CQ;
    qf.q[0] = *(const uint4*)&qp[k0];
    qf.q[1] = *(const uint4*)&qp[k0 + 16];
  }

  v8f O[16];
#pragma unroll
  for (int i = 0; i < 16; ++i)
#pragma unroll
    for (int r = 0; r < 8; ++r) O[i][r] = 0.0f;

  float mrow[8], lrow[8];
#pragma unroll
  for (int r = 0; r < 8; ++r) { mrow[r] = -3.0e38f; lrow[r] = 0.0f; }

  // ---- async stage of one 64-key tile into LDS buffer `buf` ----
  auto stage = [&](int buf, int m0) {
    async_b128(&Klds[buf][sm * KSTR + scg], &Kb[kgbase + (size_t)m0 * CQ]);
#pragma unroll
    for (int i = 0; i < 8; ++i) {
      int id = t + i * 256;
      int c = id >> 3, mg = (id & 7) * 8;
      async_b128(&Vlds[buf][c * VSTR + mg],
                 &Vb[((size_t)b * CH + c) * NPIX + m0 + mg]);
    }
  };

  stage(0, 0);   // prologue: tile 0 in flight

  for (int m0 = 0; m0 < NPIX; m0 += 64) {
    const int cur = (m0 >> 6) & 1;
    if (m0 + 64 < NPIX) {
      stage(cur ^ 1, m0 + 64);   // overlap next tile's DMA with this compute
      wait_async_le9();          // in-order: <=9 outstanding => tile `cur` done
    } else {
      wait_async_0();
    }
    __syncthreads();             // tile `cur` visible to all waves

    // ---- S = Q * K^T : 4 tiles of 16 keys ----
    v8f S[4];
#pragma unroll
    for (int kt = 0; kt < 4; ++kt) {
      Frag kf;
      const unsigned short* kp = &Klds[cur][(kt * 16 + l16) * KSTR + k0];
      kf.q[0] = *(const uint4*)&kp[0];
      kf.q[1] = *(const uint4*)&kp[16];
      v8f z;
#pragma unroll
      for (int r = 0; r < 8; ++r) z[r] = 0.0f;
      S[kt] = __builtin_amdgcn_wmma_f32_16x16x32_bf16(false, qf.v, false, kf.v,
                                                      (short)0, z, false, false);
    }

    // ---- online softmax over the 64-wide strip ----
    float mn[8], sc[8];
#pragma unroll
    for (int r = 0; r < 8; ++r) {
      float v = fmaxf(fmaxf(S[0][r], S[1][r]), fmaxf(S[2][r], S[3][r]));
#pragma unroll
      for (int off = 1; off < 16; off <<= 1)
        v = fmaxf(v, __shfl_xor(v, off, 32));
      float m2 = fmaxf(mrow[r], v);
      sc[r]   = __expf(mrow[r] - m2);
      mrow[r] = m2;
      mn[r]   = m2;
    }
    float rs[8];
#pragma unroll
    for (int r = 0; r < 8; ++r) rs[r] = 0.0f;
    unsigned short* pl = &Plds[wave][0];
#pragma unroll
    for (int kt = 0; kt < 4; ++kt)
#pragma unroll
      for (int r = 0; r < 8; ++r) {
        float p = __expf(S[kt][r] - mn[r]);
        rs[r] += p;
        pl[(r + (half ? 8 : 0)) * PSTR + kt * 16 + l16] = f2bf(p);
      }
#pragma unroll
    for (int r = 0; r < 8; ++r) {
      float s = rs[r];
#pragma unroll
      for (int off = 1; off < 16; off <<= 1) s += __shfl_xor(s, off, 32);
      lrow[r] = lrow[r] * sc[r] + s;
    }
#pragma unroll
    for (int ct = 0; ct < 16; ++ct)
#pragma unroll
      for (int r = 0; r < 8; ++r) O[ct][r] *= sc[r];

    // ---- O += P * V : P re-read from LDS in A-layout ----
#pragma unroll
    for (int mc = 0; mc < 64; mc += 32) {
      Frag pf;
      pf.q[0] = *(const uint4*)&pl[l16 * PSTR + mc + k0];
      pf.q[1] = *(const uint4*)&pl[l16 * PSTR + mc + k0 + 16];
#pragma unroll
      for (int ct = 0; ct < 16; ++ct) {
        Frag vf;
        const unsigned short* vp =
            &Vlds[cur][(ct * 16 + l16) * VSTR + mc + k0];
        vf.q[0] = *(const uint4*)&vp[0];
        vf.q[1] = *(const uint4*)&vp[16];
        O[ct] = __builtin_amdgcn_wmma_f32_16x16x32_bf16(false, pf.v, false, vf.v,
                                                        (short)0, O[ct], false, false);
      }
    }
    __syncthreads();   // all waves done reading `cur` before it is re-staged
  }

  // ---- epilogue: normalize, residual, fused store (contiguous along n) ----
  const float g = gamma[0];
  float inv[8];
#pragma unroll
  for (int r = 0; r < 8; ++r) inv[r] = 1.0f / lrow[r];
  const int qbase = qw + (half ? 8 : 0);
#pragma unroll
  for (int ct = 0; ct < 16; ++ct) {
    const int c = ct * 16 + l16;
    const size_t base = ((size_t)b * CH + c) * NPIX + qbase;
    float4 x0 = *(const float4*)&x[base];
    float4 x1 = *(const float4*)&x[base + 4];
    float4 o0, o1;
    o0.x = x0.x + g * O[ct][0] * inv[0];
    o0.y = x0.y + g * O[ct][1] * inv[1];
    o0.z = x0.z + g * O[ct][2] * inv[2];
    o0.w = x0.w + g * O[ct][3] * inv[3];
    o1.x = x1.x + g * O[ct][4] * inv[4];
    o1.y = x1.y + g * O[ct][5] * inv[5];
    o1.z = x1.z + g * O[ct][6] * inv[6];
    o1.w = x1.w + g * O[ct][7] * inv[7];
    *(float4*)&out[base]     = o0;
    *(float4*)&out[base + 4] = o1;
  }
}

// ---------------------------------------------------------------------------
extern "C" void kernel_launch(void* const* d_in, const int* in_sizes, int n_in,
                              void* d_out, int out_size, void* d_ws, size_t ws_size,
                              hipStream_t stream) {
  const float* x  = (const float*)d_in[0];
  const float* Wq = (const float*)d_in[1];
  const float* bq = (const float*)d_in[2];
  const float* Wk = (const float*)d_in[3];
  const float* bk = (const float*)d_in[4];
  const float* Wv = (const float*)d_in[5];
  const float* bv = (const float*)d_in[6];
  const float* gm = (const float*)d_in[7];

  unsigned short* Qb = (unsigned short*)d_ws;                        // 1 MB
  unsigned short* Kb = Qb + (size_t)BATCH * NPIX * CQ;               // 1 MB
  unsigned short* Vb = Kb + (size_t)BATCH * NPIX * CQ;               // 8 MB

  proj_kernel<<<dim3(NPIX / 64, 10, BATCH), 256, 0, stream>>>(
      x, Wq, bq, Wk, bk, Wv, bv, Qb, Kb, Vb);
  attn_kernel<<<dim3(NPIX / 128, BATCH), 256, 0, stream>>>(
      Qb, Kb, Vb, x, gm, (float*)d_out);
}
